// MultiQueryAttention_85229331022443
// MI455X (gfx1250) — matrix operand order repaired and checked by reading, not compiled, not run on hardware
//
#include <hip/hip_runtime.h>

// ---------------------------------------------------------------------------
// MQA forward for MI455X (gfx1250, wave32, WMMA 16x16x32 bf16).
//   B=2, S=2048, HID=2048, NH=16, HD=128.  M = B*S = 4096.
// Pipeline: f32->bf16 convert-on-stage GEMMs (Q,K,V), flash-attention with
// online softmax (all matmuls on v_wmma_f32_16x16x32_bf16), O-proj GEMM -> f32.
// CDNA5 data movement: TDM (tensor_load_to_lds, TENSORcnt) double-buffers the
// attention K tiles; async-to-LDS (ASYNCcnt) stages pure-bf16 GEMM A tiles.
// ---------------------------------------------------------------------------

typedef __attribute__((ext_vector_type(16))) __bf16 v16bf;
typedef __attribute__((ext_vector_type(8)))  float  v8f;
typedef unsigned int v4u  __attribute__((ext_vector_type(4)));
typedef int          v8i_t __attribute__((ext_vector_type(8)));
typedef int          v4i_t __attribute__((ext_vector_type(4)));

union Frag16 {
    v16bf v;
    unsigned short u[16];
};

__device__ __forceinline__ v8f wmma_bf16(v16bf a, v16bf b, v8f c) {
    // D = A(16x32 bf16) * B(32x16 bf16) + C(16x16 f32)
    return __builtin_amdgcn_wmma_f32_16x16x32_bf16(
        /*neg_a=*/false, a, /*neg_b=*/false, b,
        /*c_mod=*/(short)0, c, /*reuse_a=*/false, /*reuse_b=*/false);
}

// CDNA5 async global->LDS copy of 16 bytes (tracked by ASYNCcnt).
__device__ __forceinline__ void async_ld_b128(const __bf16* gsrc, __bf16* lds_dst) {
    unsigned int ldsoff = (unsigned int)(unsigned long long)lds_dst;  // LDS byte addr
    asm volatile("global_load_async_to_lds_b128 %0, %1, off"
                 :: "v"(ldsoff), "v"(gsrc)
                 : "memory");
}
__device__ __forceinline__ void wait_async0() {
    asm volatile("s_wait_asynccnt 0" ::: "memory");
}

// CDNA5 Tensor Data Mover: DMA a 2-D bf16 tile (tile_d0 x tile_d1 elements,
// row stride stride_d0 elements) from global memory into LDS.  One issue per
// wave; completion tracked with TENSORcnt.  D# per ISA 08_async_tensor §8.
// This toolchain exposes the 6-arg builtin: (g0, g1, g2, g3, g_extra, cpol).
__device__ __forceinline__ void tdm_load_2d_bf16(const __bf16* gaddr,
                                                 unsigned int lds_addr,
                                                 unsigned int tile_d0,
                                                 unsigned int tile_d1,
                                                 unsigned int tensor_d0,
                                                 unsigned int tensor_d1,
                                                 unsigned int stride_d0) {
    unsigned long long ga = (unsigned long long)gaddr;
    v4u g0;
    g0[0] = 1u;                                   // count=1, user mode, no gather
    g0[1] = lds_addr;                             // LDS byte address
    g0[2] = (unsigned int)ga;                     // global_addr[31:0]
    g0[3] = (unsigned int)((ga >> 32) & 0x01FFFFFFu) | (2u << 30);  // [56:32], type=2
    v8i_t g1;
    g1[0] = (int)(1u << 16);                      // wg_mask=0, data_size=1 (2 bytes)
    g1[1] = (int)((tensor_d0 & 0xFFFFu) << 16);   // tensor_dim0[15:0] @ [63:48]
    g1[2] = (int)(((tensor_d0 >> 16) & 0xFFFFu) | ((tensor_d1 & 0xFFFFu) << 16));
    g1[3] = (int)(((tensor_d1 >> 16) & 0xFFFFu) | ((tile_d0 & 0xFFFFu) << 16));
    g1[4] = (int)(tile_d1 & 0xFFFFu);             // tile_dim1; tile_dim2=0
    g1[5] = (int)stride_d0;                       // tensor_dim0_stride[31:0]
    g1[6] = 0;                                    // stride0 hi | dim1_stride lo
    g1[7] = 0;                                    // dim1_stride hi
    v4i_t z4 = {0, 0, 0, 0};
    v8i_t z8 = {0, 0, 0, 0, 0, 0, 0, 0};
    __builtin_amdgcn_tensor_load_to_lds(g0, g1, z4, z4, z8, 0);
}

// A-matrix fragment (16x32 bf16, row-major source, ld in elements).
// Order used here (same as B): lane L, M = L&15, g = L>>4 ; elem e -> K = 16g+e (contiguous).
__device__ __forceinline__ v16bf load_a_frag(const __bf16* base, int rowbase,
                                             int ld, int kbase, int lid) {
    int m = lid & 15, g = lid >> 4;
    const __bf16* p0 = base + (size_t)(rowbase + m) * ld + kbase + 16 * g;
    Frag16 f;
    *(uint4*)&f.u[0] = *(const uint4*)p0;        // K = kbase+16g   .. +8
    *(uint4*)&f.u[8] = *(const uint4*)(p0 + 8);  // K = kbase+16g+8 .. +8
    return f.v;
}

// B-matrix fragment from a transposed tile: mem layout [col][k] (B^T rows).
// Order used here (same as A): lane L, N = L&15, g = L>>4 ; elem e -> K = 16g+e (contiguous).
__device__ __forceinline__ v16bf load_b_frag_t(const __bf16* base, int colbase,
                                               int ld, int kbase, int lid) {
    int n = lid & 15, g = lid >> 4;
    const __bf16* p = base + (size_t)(colbase + n) * ld + kbase + 16 * g;
    Frag16 f;
    *(uint4*)&f.u[0] = ((const uint4*)p)[0];
    *(uint4*)&f.u[8] = ((const uint4*)p)[1];
    return f.v;
}

__device__ __forceinline__ __bf16 to_bf16(float x)  { return (__bf16)x; }
__device__ __forceinline__ __bf16 to_bf16(__bf16 x) { return x; }

// ---------------------------------------------------------------------------
// Tiled GEMM:  C[M,N] = A[M,K] @ W[K,N] + bias[N]
// A: f32 (converted on stage-in) or bf16 (async-copied).  W: f32.  C: bf16/f32.
// 256 threads = 8 waves in a 4x2 grid; each wave owns a 32x64 sub-tile
// (2 M-frags x 4 N-frags), BK = 32 per step -> 8 wmma per wave per step.
// ---------------------------------------------------------------------------
template <typename AT, typename OT>
__global__ __launch_bounds__(256) void gemm_wmma(const AT* __restrict__ A,
                                                 const float* __restrict__ W,
                                                 const float* __restrict__ bias,
                                                 OT* __restrict__ C,
                                                 int M, int N, int K) {
    __shared__ __bf16 As[128 * 32];  // [m][k], 8 KB
    __shared__ __bf16 Bs[128 * 32];  // [n][k] (W tile transposed), 8 KB

    const int tid = threadIdx.x;
    const int wave = tid >> 5, lid = tid & 31;
    const int wr = wave >> 1, wc = wave & 1;       // 4x2 wave grid
    const int mblk = blockIdx.y * 128;
    const int nblk = blockIdx.x * 128;

    v8f acc[2][4];
#pragma unroll
    for (int s = 0; s < 2; ++s)
#pragma unroll
        for (int j = 0; j < 4; ++j) acc[s][j] = (v8f){};

    for (int k0 = 0; k0 < K; k0 += 32) {
        // ---- stage A tile: 128x32, 16 elements per thread (contiguous) ----
        {
            int r = tid >> 1, c = (tid & 1) * 16;
            const AT* src = A + (size_t)(mblk + r) * K + k0 + c;
            __bf16* dst = As + r * 32 + c;
            if constexpr (sizeof(AT) == 2) {
                // pure bf16 copy -> CDNA5 async global->LDS (no VGPR round-trip)
                async_ld_b128((const __bf16*)src, dst);
                async_ld_b128((const __bf16*)src + 8, dst + 8);
            } else {
#pragma unroll
                for (int i = 0; i < 16; ++i) dst[i] = to_bf16(src[i]);
            }
            if (k0 + 32 < K)
                __builtin_prefetch(src + 32, 0, 1);  // global_prefetch next K-step
        }
        // ---- stage W tile transposed: Bs[n][k] = W[k0+k][nblk+n] ----
        {
            int kk = tid >> 3, nc = (tid & 7) * 16;
            const float* src = W + (size_t)(k0 + kk) * N + nblk + nc;
#pragma unroll
            for (int i = 0; i < 16; ++i) Bs[(size_t)(nc + i) * 32 + kk] = (__bf16)src[i];
            if (k0 + 32 < K)
                __builtin_prefetch(src + (size_t)32 * N, 0, 1);
        }
        if constexpr (sizeof(AT) == 2) wait_async0();  // own async writes done
        __syncthreads();

        v16bf a0 = load_a_frag(As, wr * 32 + 0,  32, 0, lid);
        v16bf a1 = load_a_frag(As, wr * 32 + 16, 32, 0, lid);
#pragma unroll
        for (int j = 0; j < 4; ++j) {
            v16bf b = load_b_frag_t(Bs, wc * 64 + j * 16, 32, 0, lid);
            acc[0][j] = wmma_bf16(a0, b, acc[0][j]);
            acc[1][j] = wmma_bf16(a1, b, acc[1][j]);
        }
        __syncthreads();
    }

    // ---- epilogue: bias add + store (C frag: row = r + 8g, col = lane&15) --
    const int g = lid >> 4, n = lid & 15;
#pragma unroll
    for (int s = 0; s < 2; ++s) {
#pragma unroll
        for (int j = 0; j < 4; ++j) {
            int col = nblk + wc * 64 + j * 16 + n;
            float bv = bias[col];
#pragma unroll
            for (int r = 0; r < 8; ++r) {
                int row = mblk + wr * 32 + s * 16 + r + 8 * g;
                float v = acc[s][j][r] + bv;
                C[(size_t)row * N + col] = (OT)v;
            }
        }
    }
}

// ---------------------------------------------------------------------------
// Flash attention (MQA: single shared K/V head), bf16 in, bf16 out.
// Block = one (b, head, 64-row Q tile).  8 waves in a 4x2 grid:
//   scores: wave -> 16 q-rows x 32 t-cols ;  PV: wave -> 16 q-rows x 64 d-cols.
// K tiles are DMA'd by the TDM into a double buffer: wave 0 issues the
// tensor_load_to_lds for tile t+1 before computing tile t, then waits
// TENSORcnt<=1 so the prefetch stays in flight behind the barrier.
// V tiles staged TRANSPOSED so both QK^T and PV B-fragments are contiguous
// ds_load_b128 pairs.  Online softmax state lives one-row-per-thread.
// ---------------------------------------------------------------------------
#define AS 2048
#define AHID 2048
#define AHD 128
#define ATQ 64
#define ATK 64

__global__ __launch_bounds__(256) void mqa_flash_attn(const __bf16* __restrict__ Q,
                                                      const __bf16* __restrict__ K,
                                                      const __bf16* __restrict__ V,
                                                      __bf16* __restrict__ O) {
    __shared__ __bf16 Ks[2][ATK * AHD];  // 32 KB  [t][d] double-buffered K (TDM dest)
    __shared__ __bf16 Vt[AHD * ATK];     // 16 KB  [d][t]   (= B^T for PV)
    __shared__ float  Ss[ATQ * ATK];     // 16 KB  scores f32
    __shared__ __bf16 Ps[ATQ * ATK];     //  8 KB  probs bf16
    __shared__ float  rowscale[ATQ];
    __shared__ float  rowinv[ATQ];

    const int tid = threadIdx.x;
    const int wave = tid >> 5, lid = tid & 31;
    const int wr = wave >> 1, wc = wave & 1;
    const int g = lid >> 4, n = lid & 15;
    const int b = blockIdx.z, h = blockIdx.y, q0 = blockIdx.x * ATQ;

    // Preload Q fragments for this wave's 16 rows: 4 K-steps of 32 over HD=128.
    const __bf16* Qg = Q + ((size_t)b * AS + q0) * AHID + h * AHD;
    v16bf qa[4];
#pragma unroll
    for (int kk = 0; kk < 4; ++kk)
        qa[kk] = load_a_frag(Qg, wr * 16, AHID, kk * 32, lid);

    v8f o[4];
#pragma unroll
    for (int j = 0; j < 4; ++j) o[j] = (v8f){};
    float m_i = -1e30f, l_i = 0.f;

    const __bf16* Kg = K + (size_t)b * AS * AHD;
    const __bf16* Vg = V + (size_t)b * AS * AHD;
    const float scale = 0.08838834764831845f;  // 1/sqrt(128)

    // TDM prime: wave 0 DMAs K tile 0 into buffer 0.
    if (wave == 0)
        tdm_load_2d_bf16(Kg, (unsigned int)(unsigned long long)&Ks[0][0],
                         AHD, ATK, AHD, AS, AHD);

    int buf = 0;
    for (int t0 = 0; t0 < AS; t0 += ATK) {
        // ---- TDM prefetch of next K tile into the other buffer ----
        if (wave == 0 && t0 + ATK < AS)
            tdm_load_2d_bf16(Kg + (size_t)(t0 + ATK) * AHD,
                             (unsigned int)(unsigned long long)&Ks[buf ^ 1][0],
                             AHD, ATK, AHD, AS, AHD);
        // ---- stage V tile transposed: thread owns (t-row r, 32-wide d-chunk) --
        {
            int r = tid >> 2, cc = (tid & 3) * 32;
            const __bf16* vsrc = Vg + (size_t)(t0 + r) * AHD + cc;
            __align__(16) __bf16 tmp[32];
#pragma unroll
            for (int i = 0; i < 4; ++i)
                *(uint4*)&tmp[i * 8] = ((const uint4*)vsrc)[i];
#pragma unroll
            for (int i = 0; i < 32; ++i)
                Vt[(size_t)(cc + i) * ATK + r] = tmp[i];
            if (t0 + ATK < AS)
                __builtin_prefetch(Vg + (size_t)(t0 + ATK + r) * AHD + cc, 0, 1);
        }
        // Wait for the *current* tile's DMA (leave the prefetch in flight).
        if (wave == 0) {
            if (t0 + ATK < AS) __builtin_amdgcn_s_wait_tensorcnt(1);
            else               __builtin_amdgcn_s_wait_tensorcnt(0);
        }
        __syncthreads();

        const __bf16* Kt = &Ks[buf][0];
        // ---- S = Q @ K^T : B-matrix is K^T, i.e. Kt rows are B^T rows ----
        v8f sc[2] = {(v8f){}, (v8f){}};
#pragma unroll
        for (int kk = 0; kk < 4; ++kk) {
#pragma unroll
            for (int jt = 0; jt < 2; ++jt) {
                v16bf kb = load_b_frag_t(Kt, wc * 32 + jt * 16, AHD, kk * 32, lid);
                sc[jt] = wmma_bf16(qa[kk], kb, sc[jt]);
            }
        }
#pragma unroll
        for (int jt = 0; jt < 2; ++jt)
#pragma unroll
            for (int r = 0; r < 8; ++r)
                Ss[(wr * 16 + r + 8 * g) * ATK + wc * 32 + jt * 16 + n] =
                    sc[jt][r] * scale;
        __syncthreads();

        // ---- online softmax: one thread per q-row ----
        if (tid < ATQ) {
            float m_new = m_i;
#pragma unroll 8
            for (int j = 0; j < ATK; ++j) m_new = fmaxf(m_new, Ss[tid * ATK + j]);
            float rs = __expf(m_i - m_new);
            float sum = 0.f;
#pragma unroll 8
            for (int j = 0; j < ATK; ++j) {
                float p = __expf(Ss[tid * ATK + j] - m_new);
                sum += p;
                Ps[tid * ATK + j] = (__bf16)p;
            }
            l_i = l_i * rs + sum;
            m_i = m_new;
            rowscale[tid] = rs;
        }
        __syncthreads();

        // ---- rescale accumulator, then O += P @ V (Vt rows are B^T rows) --
#pragma unroll
        for (int r = 0; r < 8; ++r) {
            float f = rowscale[wr * 16 + r + 8 * g];
#pragma unroll
            for (int j = 0; j < 4; ++j) o[j][r] *= f;
        }
#pragma unroll
        for (int kk = 0; kk < 2; ++kk) {
            v16bf pa = load_a_frag(Ps, wr * 16, ATK, kk * 32, lid);
#pragma unroll
            for (int jt = 0; jt < 4; ++jt) {
                v16bf vb = load_b_frag_t(Vt, wc * 64 + jt * 16, ATK, kk * 32, lid);
                o[jt] = wmma_bf16(pa, vb, o[jt]);
            }
        }
        __syncthreads();  // protect Vt/Ss/Ps and free Ks[buf] before next stage-in
        buf ^= 1;
    }

    if (tid < ATQ) rowinv[tid] = 1.0f / l_i;
    __syncthreads();

    // ---- normalize + store: attn[b, s, h, d] packed as [B,S,HID] ----
#pragma unroll
    for (int r = 0; r < 8; ++r) {
        int row = wr * 16 + r + 8 * g;
        float inv = rowinv[row];
#pragma unroll
        for (int jt = 0; jt < 4; ++jt) {
            int col = h * AHD + wc * 64 + jt * 16 + n;
            O[((size_t)b * AS + q0 + row) * AHID + col] = (__bf16)(o[jt][r] * inv);
        }
    }
}

// ---------------------------------------------------------------------------
// Host launcher
// ---------------------------------------------------------------------------
extern "C" void kernel_launch(void* const* d_in, const int* in_sizes, int n_in,
                              void* d_out, int out_size, void* d_ws, size_t ws_size,
                              hipStream_t stream) {
    const float* hs = (const float*)d_in[0];
    const float* Wq = (const float*)d_in[1];
    const float* bq = (const float*)d_in[2];
    const float* Wk = (const float*)d_in[3];
    const float* bk = (const float*)d_in[4];
    const float* Wv = (const float*)d_in[5];
    const float* bv = (const float*)d_in[6];
    const float* Wo = (const float*)d_in[7];
    const float* bo = (const float*)d_in[8];
    float* out = (float*)d_out;

    const int M = 2 * 2048;   // B*S
    const int N = 2048;       // HID
    const int K = 2048;       // HID
    const int HD = 128;

    char* ws = (char*)d_ws;
    __bf16* Qb = (__bf16*)(ws);                                // 16 MiB
    __bf16* Kb = (__bf16*)(ws + (size_t)16 * 1024 * 1024);     //  1 MiB
    __bf16* Vb = (__bf16*)(ws + (size_t)17 * 1024 * 1024);     //  1 MiB
    __bf16* Ab = (__bf16*)(ws + (size_t)18 * 1024 * 1024);     // 16 MiB

    dim3 blk(256);
    // Projections (bf16 outputs)
    gemm_wmma<float, __bf16><<<dim3(N / 128, M / 128), blk, 0, stream>>>(hs, Wq, bq, Qb, M, N, K);
    gemm_wmma<float, __bf16><<<dim3(HD / 128, M / 128), blk, 0, stream>>>(hs, Wk, bk, Kb, M, HD, K);
    gemm_wmma<float, __bf16><<<dim3(HD / 128, M / 128), blk, 0, stream>>>(hs, Wv, bv, Vb, M, HD, K);
    // Flash attention: (q-tiles, heads, batch)
    mqa_flash_attn<<<dim3(2048 / 64, 16, 2), blk, 0, stream>>>(Qb, Kb, Vb, Ab);
    // Output projection (f32 output)
    gemm_wmma<__bf16, float><<<dim3(16, 32), blk, 0, stream>>>(Ab, Wo, bo, out, M, N, K);
}
